// M2oE_61478161875320
// MI455X (gfx1250) — compile-verified, matrix operand order checked by reading, not run on hardware
//
#include <hip/hip_runtime.h>
#include <hip/hip_bf16.h>
#include <math.h>

// ---------------------------------------------------------------------------
// Switch-MoE forward (torch-faithful semantics) for MI455X / gfx1250, wave32.
// B=4, N=512, D=1024, E=8, H=4096. Only 32 (b, n<8) rows ever produce output,
// and only expert 0 is ever used -> two tiny M=32 WMMA GEMMs streaming the
// expert-0 weights once (bandwidth-bound: ~32MB weights + 8MB zero + 8MB out).
// This round: activation panels staged into LDS via the Tensor Data Mover
// (tensor_load_to_lds + s_wait_tensorcnt) and read back as ds_load_b128.
// ---------------------------------------------------------------------------

typedef __attribute__((ext_vector_type(16))) _Float16     v16h;
typedef __attribute__((ext_vector_type(8)))  _Float16     v8h;
typedef __attribute__((ext_vector_type(8)))  float        v8f;
typedef __attribute__((ext_vector_type(4)))  unsigned int u32x4;
typedef __attribute__((ext_vector_type(8)))  int          i32x8;
typedef __attribute__((ext_vector_type(4)))  int          i32x4;

#define BB   4
#define NN   512
#define DD   1024
#define EE   8
#define HH   4096
#define TOK  (BB * NN)           // 2048
#define OUTN (BB * NN * DD)      // 2097152
#define MROWS 32                 // compact rows: r = b*8 + n  (n < 8)

#if defined(__HIP_DEVICE_COMPILE__) && __has_builtin(__builtin_amdgcn_tensor_load_to_lds)
#define USE_TDM 1
#else
#define USE_TDM 0
#endif

// ---------------------------------------------------------------------------
// TDM: issue a 2D f16 tile load Global->LDS and wait on TENSORcnt.
// D# packing per CDNA5 ISA:
//   group0: count=1 | lds_addr[63:32] | global_addr[120:64] | type=2[127:126]
//   group1: data_size=1(2B)[17:16], tensor_dim0[79:48], tensor_dim1[111:80],
//           tile_dim0[127:112], tile_dim1[143:128], dim0_stride[207:160]
// groups 2/3 zero (2-D tensor -> VADDR2/3 effectively unused).
// ---------------------------------------------------------------------------
__device__ __forceinline__ void tdm_load_tile_f16(unsigned int lds_byte,
                                                  const _Float16* gptr,
                                                  unsigned int dim0,    // elements
                                                  unsigned int dim1,    // rows
                                                  unsigned int stride0) // elements
{
#if USE_TDM
    unsigned long long ga = (unsigned long long)(size_t)gptr;
    u32x4 g0;
    g0[0] = 1u;                                            // count=1, user desc
    g0[1] = lds_byte;                                      // lds_addr
    g0[2] = (unsigned int)ga;                              // global_addr[31:0]
    g0[3] = (unsigned int)((ga >> 32) & 0x01ffffffu) | (2u << 30); // [56:32]|type=2
    i32x8 g1;
    g1[0] = 0x00010000;                                    // data_size=1 (2 bytes)
    g1[1] = (int)((dim0 & 0xffffu) << 16);                 // tensor_dim0[15:0]
    g1[2] = (int)(((dim0 >> 16) & 0xffffu) | ((dim1 & 0xffffu) << 16));
    g1[3] = (int)(((dim1 >> 16) & 0xffffu) | ((dim0 & 0xffffu) << 16)); // tile_dim0
    g1[4] = (int)(dim1 & 0xffffu);                         // tile_dim1 (tile_dim2=0)
    g1[5] = (int)stride0;                                  // dim0_stride[31:0]
    g1[6] = 0;
    g1[7] = 0;
    i32x4 z4 = {};
#if defined(__clang_major__) && (__clang_major__ >= 23)
    i32x8 z8 = {};
    __builtin_amdgcn_tensor_load_to_lds(g0, g1, z4, z4, z8, 0);
#else
    __builtin_amdgcn_tensor_load_to_lds(g0, g1, z4, z4, 0);
#endif
    __builtin_amdgcn_s_wait_tensorcnt(0);
#else
    (void)lds_byte; (void)gptr; (void)dim0; (void)dim1; (void)stride0;
#endif
}

// ---------------------------------------------------------------------------
// Kernel 1: gating. One block (8 waves) per token; wave e computes logit_e.
// ---------------------------------------------------------------------------
__global__ __launch_bounds__(256) void gate_kernel(const float* __restrict__ x,
                                                   const float* __restrict__ wg,
                                                   const float* __restrict__ bg,
                                                   float* __restrict__ gate,
                                                   int* __restrict__ topidx) {
    const int tok  = blockIdx.x;
    const int lane = threadIdx.x & 31;
    const int e    = threadIdx.x >> 5;     // 8 waves -> 8 experts
    __shared__ float slog[EE];

    const float* xr = x + (size_t)tok * DD;
    float s = 0.f;
    for (int k = lane; k < DD; k += 32)
        s += xr[k] * wg[k * EE + e];
    #pragma unroll
    for (int off = 16; off > 0; off >>= 1)
        s += __shfl_xor(s, off);
    if (lane == 0) slog[e] = s + bg[e];
    __syncthreads();

    if (threadIdx.x == 0) {
        float mx = slog[0]; int am = 0;
        #pragma unroll
        for (int j = 1; j < EE; ++j)
            if (slog[j] > mx) { mx = slog[j]; am = j; }   // first-max, like argmax
        float ex[EE], sum = 0.f;
        #pragma unroll
        for (int j = 0; j < EE; ++j) { ex[j] = __expf(slog[j] - mx); sum += ex[j]; }
        const float inv = 1.f / sum;
        #pragma unroll
        for (int j = 0; j < EE; ++j) gate[(size_t)tok * EE + j] = ex[j] * inv;
        topidx[tok] = am;
    }
}

// ---------------------------------------------------------------------------
// Kernel 2: gate_scores (32 values) + aux loss. Single block.
// mask[b][n] = 1 iff exists ni with top_idx[b,ni] == n   (n < 8, e = 0 only)
// ---------------------------------------------------------------------------
__global__ __launch_bounds__(256) void scores_loss_kernel(const float* __restrict__ gate,
                                                          const int* __restrict__ topidx,
                                                          float* __restrict__ gs32,
                                                          float* __restrict__ loss_out) {
    __shared__ int   smask[MROWS];
    __shared__ float smasked[MROWS];
    __shared__ float sden[EE];
    __shared__ float sgs[MROWS];
    const int tid = threadIdx.x;

    if (tid < MROWS) smask[tid] = 0;
    __syncthreads();

    for (int t = tid; t < TOK; t += blockDim.x) {
        int v = topidx[t];                 // in [0,8)
        int b = t >> 9;                    // N = 512
        smask[b * EE + v] = 1;             // benign race: all write 1
    }
    __syncthreads();

    if (tid < MROWS) {
        int b = tid >> 3, n = tid & 7;
        float g = gate[((size_t)(b * NN + n)) * EE + 0];
        smasked[tid] = g * (float)smask[tid];
    }
    __syncthreads();

    if (tid < EE) {
        float d = 1e-6f;
        #pragma unroll
        for (int b = 0; b < BB; ++b) d += smasked[b * EE + tid];
        sden[tid] = d;
    }
    __syncthreads();

    if (tid < MROWS) {
        int n = tid & 7;
        float gs = smasked[tid] / sden[n] * 4.0f;   // capacity = int(1.0 * B) = 4
        sgs[tid]  = gs;
        gs32[tid] = gs;
    }
    __syncthreads();

    if (tid == 0) {
        float S1i = 0.f, S2i = 0.f, S1l = 0.f, S2l = 0.f;
        #pragma unroll
        for (int n = 0; n < EE; ++n) {
            float imp = 0.f, ld = 0.f;
            #pragma unroll
            for (int b = 0; b < BB; ++b) {
                float g = sgs[b * EE + n];
                imp += g;
                ld  += (g > 0.f) ? 1.f : 0.f;
            }
            S1i += imp; S2i += imp * imp;
            S1l += ld;  S2l += ld * ld;
        }
        const float CNT = (float)(NN * EE);         // 4096 entries in [N,E]
        float mi = S1i / CNT, ml = S1l / CNT;
        float vi = (S2i - CNT * mi * mi) / (CNT - 1.f);   // ddof=1
        float vl = (S2l - CNT * ml * ml) / (CNT - 1.f);
        loss_out[0] = vi / (mi * mi + 1e-10f) + vl / (ml * ml + 1e-10f);
    }
}

// ---------------------------------------------------------------------------
// Kernel 3: zero the 8 MB output (everything outside the 32 live rows is 0).
// ---------------------------------------------------------------------------
__global__ __launch_bounds__(256) void zero_out_kernel(float4* __restrict__ o) {
    int i = blockIdx.x * blockDim.x + threadIdx.x;
    if (i < OUTN / 4) o[i] = make_float4(0.f, 0.f, 0.f, 0.f);
}

// ---------------------------------------------------------------------------
// Kernel 4: pack the 32 live token rows of x into a compact f16 [32,1024].
// ---------------------------------------------------------------------------
__global__ __launch_bounds__(256) void pack_x_kernel(const float* __restrict__ x,
                                                     _Float16* __restrict__ xf) {
    int i = blockIdx.x * blockDim.x + threadIdx.x;   // 32*1024 elements
    if (i < MROWS * DD) {
        int r = i >> 10, k = i & (DD - 1);
        int b = r >> 3, n = r & 7;
        xf[i] = (_Float16)x[((size_t)(b * NN + n)) * DD + k];
    }
}

// ---------------------------------------------------------------------------
// WMMA fragment loaders (layouts from CDNA5 ISA §7.12.2).
// A 16x32 f16: lane row M = lane&15; half j -> K = kb + j + 8*((j>=8)+(lane>=16))
//   -> two contiguous 16-byte loads per lane from a row-major f16 panel (LDS).
// B 32x16 f16: lane col N = lane&15; same K mapping; gathered from row-major
//   f32 weights (each j-load coalesces across lanes: 16 consecutive floats).
// ---------------------------------------------------------------------------
__device__ __forceinline__ v16h load_a_frag(const _Float16* Xf,
                                            int row, int ld, int kb, int lane) {
    int off = row * ld + kb + ((lane >> 4) << 3);
    v8h lo = *(const v8h*)(Xf + off);
    v8h hi = *(const v8h*)(Xf + off + 16);
    v16h a;
    #pragma unroll
    for (int j = 0; j < 8; ++j) { a[j] = lo[j]; a[j + 8] = hi[j]; }
    return a;
}

__device__ __forceinline__ v16h load_b_frag(const float* __restrict__ W,
                                            int ldn, int kb, int col, int lane) {
    int k0 = kb + ((lane >> 4) << 3);
    const float* p = W + (size_t)k0 * ldn + col;
    v16h b;
    #pragma unroll
    for (int j = 0; j < 8; ++j) b[j]     = (_Float16)p[(size_t)j * ldn];
    #pragma unroll
    for (int j = 0; j < 8; ++j) b[j + 8] = (_Float16)p[(size_t)(j + 16) * ldn];
    return b;
}

__device__ __forceinline__ float gelu_exact(float v) {
    return 0.5f * v * (1.0f + erff(v * 0.70710678118654752440f));
}

// ---------------------------------------------------------------------------
// Kernel 5: GEMM1 + bias + exact GELU.  [32,1024]f16 x [1024,4096] -> h f16.
// 8 waves/block, wave owns 16 columns; 32 blocks cover H=4096.
// A panel (64 KB) staged once into LDS by the TDM; fragments via ds_load_b128.
// ---------------------------------------------------------------------------
__global__ __launch_bounds__(256) void ffn_gemm1_kernel(const _Float16* __restrict__ xf,
                                                        const float* __restrict__ w1,
                                                        const float* __restrict__ b1,
                                                        _Float16* __restrict__ hb) {
    __shared__ __align__(16) _Float16 sA[MROWS * DD];   // 64 KB
    const int lane = threadIdx.x & 31;
    const int wave = threadIdx.x >> 5;
    const int col  = blockIdx.x * 128 + wave * 16 + (lane & 15);

#if USE_TDM
    if (wave == 0)
        tdm_load_tile_f16((unsigned int)(size_t)&sA[0], xf, DD, MROWS, DD);
    __syncthreads();
#else
    for (int i = threadIdx.x * 8; i < MROWS * DD; i += 256 * 8)
        *(v8h*)&sA[i] = *(const v8h*)&xf[i];
    __syncthreads();
#endif

    v8f c0 = {}; v8f c1 = {};
    for (int kb = 0; kb < DD; kb += 32) {
        if (kb + 32 < DD)
            __builtin_prefetch(w1 + (size_t)(kb + 32) * HH + col, 0, 1);
        v16h a0 = load_a_frag(sA, (lane & 15),      DD, kb, lane);
        v16h a1 = load_a_frag(sA, (lane & 15) + 16, DD, kb, lane);
        v16h bf = load_b_frag(w1, HH, kb, col, lane);
        c0 = __builtin_amdgcn_wmma_f32_16x16x32_f16(false, a0, false, bf,
                                                    (short)0, c0, false, false);
        c1 = __builtin_amdgcn_wmma_f32_16x16x32_f16(false, a1, false, bf,
                                                    (short)0, c1, false, false);
    }

    const float bias = b1[col];
    #pragma unroll
    for (int r = 0; r < 8; ++r) {
        int m = r + ((lane >> 4) << 3);          // C/D layout: row = r + 8*(lane>=16)
        hb[(size_t)m * HH + col]        = (_Float16)gelu_exact(c0[r] + bias);
        hb[(size_t)(m + 16) * HH + col] = (_Float16)gelu_exact(c1[r] + bias);
    }
}

// ---------------------------------------------------------------------------
// Kernel 6: GEMM2 + bias + gate-score scale, scattered into d_out rows.
// [32,4096]f16 x [4096,1024] -> out rows (b, n<8). 8 blocks cover D=1024.
// h panel staged in four 64 KB K-chunks via TDM (tile 1024x32, stride 4096).
// ---------------------------------------------------------------------------
__global__ __launch_bounds__(256) void ffn_gemm2_kernel(const _Float16* __restrict__ hb,
                                                        const float* __restrict__ w2,
                                                        const float* __restrict__ b2,
                                                        const float* __restrict__ gs32,
                                                        float* __restrict__ out) {
    __shared__ __align__(16) _Float16 sA[MROWS * 1024]; // 64 KB chunk
    const int lane = threadIdx.x & 31;
    const int wave = threadIdx.x >> 5;
    const int col  = blockIdx.x * 128 + wave * 16 + (lane & 15);

    v8f c0 = {}; v8f c1 = {};
    for (int chunk = 0; chunk < HH / 1024; ++chunk) {
        __syncthreads();                    // previous chunk fully consumed
#if USE_TDM
        if (wave == 0)
            tdm_load_tile_f16((unsigned int)(size_t)&sA[0],
                              hb + (size_t)chunk * 1024, 1024, MROWS, HH);
        __syncthreads();
#else
        for (int i = threadIdx.x; i < MROWS * 1024 / 8; i += 256) {
            int r = i >> 7, c = (i & 127) * 8;
            *(v8h*)&sA[r * 1024 + c] =
                *(const v8h*)&hb[(size_t)r * HH + chunk * 1024 + c];
        }
        __syncthreads();
#endif
        const int kglob0 = chunk * 1024;
        for (int kb = 0; kb < 1024; kb += 32) {
            int kg = kglob0 + kb;
            if (kg + 32 < HH)
                __builtin_prefetch(w2 + (size_t)(kg + 32) * DD + col, 0, 1);
            v16h a0 = load_a_frag(sA, (lane & 15),      1024, kb, lane);
            v16h a1 = load_a_frag(sA, (lane & 15) + 16, 1024, kb, lane);
            v16h bf = load_b_frag(w2, DD, kg, col, lane);
            c0 = __builtin_amdgcn_wmma_f32_16x16x32_f16(false, a0, false, bf,
                                                        (short)0, c0, false, false);
            c1 = __builtin_amdgcn_wmma_f32_16x16x32_f16(false, a1, false, bf,
                                                        (short)0, c1, false, false);
        }
    }

    const float bias = b2[col];
    #pragma unroll
    for (int r = 0; r < 8; ++r) {
        int m0 = r + ((lane >> 4) << 3);
        int m1 = m0 + 16;
        int t0 = ((m0 >> 3) * NN) + (m0 & 7);    // token = b*512 + n
        int t1 = ((m1 >> 3) * NN) + (m1 & 7);
        out[(size_t)t0 * DD + col] = gs32[m0] * (c0[r] + bias);
        out[(size_t)t1 * DD + col] = gs32[m1] * (c1[r] + bias);
    }
}

// ---------------------------------------------------------------------------
// Launch
// ---------------------------------------------------------------------------
extern "C" void kernel_launch(void* const* d_in, const int* in_sizes, int n_in,
                              void* d_out, int out_size, void* d_ws, size_t ws_size,
                              hipStream_t stream) {
    const float* x  = (const float*)d_in[0];
    const float* wg = (const float*)d_in[1];
    const float* bg = (const float*)d_in[2];
    const float* w1 = (const float*)d_in[3];   // [E,D,H]; expert 0 at offset 0
    const float* b1 = (const float*)d_in[4];   // [E,H]
    const float* w2 = (const float*)d_in[5];   // [E,H,D]
    const float* b2 = (const float*)d_in[6];   // [E,D]
    float* out = (float*)d_out;

    char* ws = (char*)d_ws;
    float*    gate_ws = (float*)(ws);                          // 16384 f32 = 64 KB
    int*      top_ws  = (int*)  (ws + 65536);                  // 2048 i32  =  8 KB
    float*    gs_ws   = (float*)(ws + 73728);                  //   32 f32
    _Float16* xf16    = (_Float16*)(ws + 73856);               // 32x1024 f16 = 64 KB
    _Float16* hb16    = (_Float16*)(ws + 139392);              // 32x4096 f16 = 256 KB

    gate_kernel<<<TOK, 256, 0, stream>>>(x, wg, bg, gate_ws, top_ws);
    scores_loss_kernel<<<1, 256, 0, stream>>>(gate_ws, top_ws, gs_ws, out + OUTN);
    zero_out_kernel<<<(OUTN / 4 + 255) / 256, 256, 0, stream>>>((float4*)out);
    pack_x_kernel<<<(MROWS * DD + 255) / 256, 256, 0, stream>>>(x, xf16);
    ffn_gemm1_kernel<<<HH / 128, 256, 0, stream>>>(xf16, w1, b1, hb16);
    ffn_gemm2_kernel<<<DD / 128, 256, 0, stream>>>(hb16, w2, b2, gs_ws, out);
}
